// FDSM_40295383171690
// MI455X (gfx1250) — matrix operand order repaired and checked by reading, not compiled, not run on hardware
//
#include <hip/hip_runtime.h>
#include <math.h>

// ---------------- problem constants ----------------
#define B_   8
#define C_   64
#define H_   128
#define W_   128
#define F_   4
#define WF_  65
#define HW_  (H_ * W_)          // 16384
#define PITCH 129               // padded LDS row pitch (float2) to dodge bank conflicts
#define KP   72                 // bf16 tile row pitch: 144 B, 16-B aligned rows

typedef __bf16 bf16_t;
typedef __attribute__((ext_vector_type(16))) __bf16 v16bf;
typedef __attribute__((ext_vector_type(8)))  __bf16 v8bf;
typedef __attribute__((ext_vector_type(8)))  float  v8f;

// ---------------- CDNA5 async-to-LDS feature probe ----------------
#if defined(__has_builtin)
#  if __has_builtin(__builtin_amdgcn_global_load_async_to_lds_b128) && \
      __has_builtin(__builtin_amdgcn_s_wait_asynccnt)
#    define HAVE_ASYNC_LDS 1
#    warning "CDNA5 probe: async-to-LDS builtins AVAILABLE - async path compiled"
#  endif
#endif
#ifndef HAVE_ASYNC_LDS
#  define HAVE_ASYNC_LDS 0
#  warning "CDNA5 probe: async-to-LDS builtins NOT FOUND - synchronous load path compiled"
#endif

#if HAVE_ASYNC_LDS
typedef int v4i_ __attribute__((vector_size(4 * sizeof(int))));
typedef __attribute__((address_space(1))) v4i_ g_v4i;
typedef __attribute__((address_space(3))) v4i_ l_v4i;
__device__ __forceinline__ g_v4i* to_glob_v4(const void* p) {
    return (g_v4i*)(unsigned long long)p;
}
__device__ __forceinline__ l_v4i* to_lds_v4(void* p) {
    return (l_v4i*)(unsigned int)(unsigned long long)p;
}
#endif

// ===================================================================
// K0: zero scratch accumulator
// ===================================================================
__global__ void zero_kernel(float* p, int n) {
    int i = blockIdx.x * blockDim.x + threadIdx.x;
    if (i < n) p[i] = 0.f;
}

// ===================================================================
// K1: group-norm stats. concat(x,x) with 32 groups == 16 groups over x.
// ===================================================================
__global__ __launch_bounds__(256)
void gn_stats_kernel(const float* __restrict__ feat,
                     float* __restrict__ mu, float* __restrict__ rsig)
{
    const int bg = blockIdx.x;            // b*16 + g
    const int b  = bg >> 4, g = bg & 15;
    const float* p = feat + ((size_t)b * C_ + (size_t)g * 4) * HW_;

    float s1 = 0.f, s2 = 0.f;
    for (int i = threadIdx.x; i < 4 * HW_; i += 256) {
        float x = p[i];
        s1 += x; s2 += x * x;
    }
    for (int o = 16; o > 0; o >>= 1) {
        s1 += __shfl_down(s1, o);
        s2 += __shfl_down(s2, o);
    }
    __shared__ float r1[8], r2[8];
    const int wid = threadIdx.x >> 5;
    if ((threadIdx.x & 31) == 0) { r1[wid] = s1; r2[wid] = s2; }
    __syncthreads();
    if (threadIdx.x == 0) {
        float a = 0.f, q = 0.f;
        for (int i = 0; i < 8; ++i) { a += r1[i]; q += r2[i]; }
        const float inv = 1.0f / (float)(4 * HW_);
        float m = a * inv;
        float v = q * inv - m * m;
        mu[bg]   = m;
        rsig[bg] = rsqrtf(v + 1e-5f);
    }
}

// ===================================================================
// K2: fold gamma/beta + concat into effective 64x64 weight (bf16,
// K-major in global scratch) and effective bias.
// ===================================================================
__global__ void prep_weights_kernel(const float* __restrict__ agg_w,  // [64][128]
                                    const float* __restrict__ agg_b,
                                    const float* __restrict__ gamma,  // [128]
                                    const float* __restrict__ beta,   // [128]
                                    bf16_t* __restrict__ Bw,          // [64(k)][64(n)]
                                    float*  __restrict__ beff)
{
    const int o = threadIdx.x;
    if (o >= 64) return;
    float bias = agg_b[o];
    for (int c = 0; c < 128; ++c) bias += agg_w[o * 128 + c] * beta[c];
    beff[o] = bias;
    for (int c = 0; c < 64; ++c) {
        float we = agg_w[o * 128 + c] * gamma[c]
                 + agg_w[o * 128 + 64 + c] * gamma[64 + c];
        Bw[c * 64 + o] = (bf16_t)we;   // B[k=c][n=o]
    }
}

// ===================================================================
// K3: gating GEMM, wave-specialized; fragments assembled with aligned
// 16-byte LDS vector loads + shufflevector (no stack round-trips).
// Layouts per CDNA5 ISA:
//   A 16x32 16-bit: elem e <-> K = kc*32 + khA + e (+8 for e>=8)
//   B 32x16 16-bit: elem e <-> K = kc*32 + khB + e (+8 for e>=8)   (B held as [n][k])
// ===================================================================
__global__ __launch_bounds__(256)
void gate_gemm_kernel(const float*  __restrict__ feat,
                      const float*  __restrict__ mu,
                      const float*  __restrict__ rsig,
                      const bf16_t* __restrict__ Bw,
                      const float*  __restrict__ beff,
                      float*        __restrict__ pooled)
{
    __shared__ alignas(16) bf16_t Bsh[64][KP];      // [n][k]
    __shared__ alignas(16) bf16_t Ash[2][16][KP];   // [subtile][m][k]
    __shared__ float  biasSh[64];
    __shared__ float  muSh[16], rsSh[16];

    const int tid  = threadIdx.x;
    const int lane = tid & 31;
    const int wid  = tid >> 5;
    const int nt   = wid & 3;      // N-tile owned by this wave
    const int st   = wid >> 2;     // pixel-subtile owned by this wave
    const int b    = blockIdx.y;

    for (int i = tid; i < 64 * 64; i += 256) {
        int k = i >> 6, nn = i & 63;
        Bsh[nn][k] = Bw[i];                        // transpose to [n][k]
    }
    if (tid < 64) biasSh[tid] = beff[tid];
    if (tid < 16) { muSh[tid] = mu[b * 16 + tid]; rsSh[tid] = rsig[b * 16 + tid]; }
    __syncthreads();

    const int n   = lane & 15;
    const int khB = (lane < 16) ? 0 : 8;
    v16bf bfrag[2];
#pragma unroll
    for (int kc = 0; kc < 2; ++kc) {
        const bf16_t* row = &Bsh[nt * 16 + n][0];
        v8bf lo = *(const v8bf*)&row[kc * 32 + khB];       // K khB..khB+7
        v8bf hi = *(const v8bf*)&row[kc * 32 + 16 + khB];  // K 16+khB..16+khB+7
        bfrag[kc] = __builtin_shufflevector(lo, hi,
                        0, 1, 2, 3, 4, 5, 6, 7, 8, 9, 10, 11, 12, 13, 14, 15);
    }

    const int   m   = lane & 15;
    const int   khA = (lane < 16) ? 0 : 8;
    const float bn  = biasSh[nt * 16 + n];

    float acc = 0.f;
    const float* fb = feat + (size_t)b * C_ * HW_;

    for (int t0 = blockIdx.x * 2; t0 < HW_ / 16; t0 += gridDim.x * 2) {
        __syncthreads();   // previous iteration's readers done before rewrite
        for (int i = tid; i < 2 * 16 * 64; i += 256) {
            int tt = i >> 10;
            int mm = i & 15;
            int c  = (i >> 4) & 63;
            float x = fb[(size_t)c * HW_ + (t0 + tt) * 16 + mm];
            Ash[tt][mm][c] = (bf16_t)((x - muSh[c >> 2]) * rsSh[c >> 2]);
        }
        __syncthreads();

        v16bf afrag[2];
#pragma unroll
        for (int kc = 0; kc < 2; ++kc) {
            const bf16_t* row = &Ash[st][m][0];
            v8bf lo = *(const v8bf*)&row[kc * 32 + khA];        // K kc*32+khA..+7
            v8bf hi = *(const v8bf*)&row[kc * 32 + 16 + khA];   // K kc*32+16+khA..+7
            afrag[kc] = __builtin_shufflevector(lo, hi,
                            0, 1, 2, 3, 4, 5, 6, 7, 8, 9, 10, 11, 12, 13, 14, 15);
        }

        v8f cfrag = {0.f, 0.f, 0.f, 0.f, 0.f, 0.f, 0.f, 0.f};
        cfrag = __builtin_amdgcn_wmma_f32_16x16x32_bf16(
                    false, afrag[0], false, bfrag[0], (short)0, cfrag, false, false);
        cfrag = __builtin_amdgcn_wmma_f32_16x16x32_bf16(
                    false, afrag[1], false, bfrag[1], (short)0, cfrag, false, false);

        float s = 0.f;
#pragma unroll
        for (int r = 0; r < 8; ++r) s += fmaxf(cfrag[r] + bn, 0.f);
        acc += s;
    }

    acc += __shfl_xor(acc, 16);        // combine the two M-halves of column N
    if (lane < 16)
        atomicAdd(&pooled[b * 64 + nt * 16 + n], acc);
}

// ===================================================================
// K4: pooled -> logits (4x64 FC) -> softmax gate weights
// ===================================================================
__global__ void gate_softmax_kernel(const float* __restrict__ pooled,
                                    const float* __restrict__ wg_w,
                                    const float* __restrict__ wg_b,
                                    float* __restrict__ gatew)
{
    __shared__ float red[4][64];
    __shared__ float lg[4];
    const int b = blockIdx.x, tid = threadIdx.x;
    float pm = pooled[b * 64 + tid] * (1.0f / (float)HW_);
    for (int f = 0; f < 4; ++f) red[f][tid] = pm * wg_w[f * 64 + tid];
    __syncthreads();
    if (tid < 4) {
        float s = wg_b[tid];
        for (int c = 0; c < 64; ++c) s += red[tid][c];
        lg[tid] = s;
    }
    __syncthreads();
    if (tid == 0) {
        float mx = fmaxf(fmaxf(lg[0], lg[1]), fmaxf(lg[2], lg[3]));
        float e0 = __expf(lg[0] - mx), e1 = __expf(lg[1] - mx);
        float e2 = __expf(lg[2] - mx), e3 = __expf(lg[3] - mx);
        float inv = 1.0f / (e0 + e1 + e2 + e3);
        gatew[b * 4 + 0] = e0 * inv; gatew[b * 4 + 1] = e1 * inv;
        gatew[b * 4 + 2] = e2 * inv; gatew[b * 4 + 3] = e3 * inv;
    }
}

// ===================================================================
// Batched radix-2 FFT: 128 independent length-128 transforms in LDS.
// ===================================================================
__device__ __forceinline__
void fft128_batch(float2* __restrict__ S, const float2* __restrict__ TW,
                  int tmul, int emul, int inverse, int tid)
{
    for (int i = tid; i < 128 * 128; i += 256) {
        int t = i >> 7, e = i & 127;
        int r = (int)(__brev((unsigned)e) >> 25);
        if (e < r) {
            float2* p = S + t * tmul;
            float2 a = p[e * emul];
            p[e * emul] = p[r * emul];
            p[r * emul] = a;
        }
    }
    __syncthreads();

    for (int s = 1; s <= 7; ++s) {
        const int half = 1 << (s - 1);
        for (int i = tid; i < 128 * 64; i += 256) {
            int t = i >> 6, j = i & 63;
            int pos = j & (half - 1);
            int k   = ((j >> (s - 1)) << s) + pos;
            float2 w = TW[pos << (7 - s)];
            float wy = inverse ? -w.y : w.y;
            float2* p = S + t * tmul;
            float2 a = p[k * emul];
            float2 q = p[(k + half) * emul];
            float br = q.x * w.x - q.y * wy;
            float bi = q.x * wy + q.y * w.x;
            p[k * emul]          = make_float2(a.x + br, a.y + bi);
            p[(k + half) * emul] = make_float2(a.x - br, a.y - bi);
        }
        __syncthreads();
    }
}

// ===================================================================
// K5: fused spectral kernel, one block per (b,c) image.
// ===================================================================
__global__ __launch_bounds__(256)
void spectral_kernel(const float* __restrict__ feat,
                     const float* __restrict__ ds_w,   // [F][C][H][WF][2]
                     const float* __restrict__ gatew,  // [B][F]
                     const float* __restrict__ rw_p,   // [1]
                     float* __restrict__ out)
{
    extern __shared__ float2 sm[];
    float2* S  = sm;                  // [128][PITCH]
    float2* TW = sm + 128 * PITCH;    // [64] twiddles

    const int tid = threadIdx.x;
    const int bc  = blockIdx.x;
    const int b   = bc >> 6;
    const int c   = bc & 63;

    if (tid < 64) {
        float sn, cs;
        __sincosf(-3.14159265358979323846f * (float)tid / 64.0f, &sn, &cs);
        TW[tid] = make_float2(cs, sn);
    }
    const float wt0 = gatew[b * 4 + 0], wt1 = gatew[b * 4 + 1];
    const float wt2 = gatew[b * 4 + 2], wt3 = gatew[b * 4 + 3];
    const float rw = rw_p[0];

    const float* img = feat + (size_t)bc * HW_;
    const float* dwc = ds_w + (size_t)c * (H_ * WF_ * 2);
    const size_t fs  = (size_t)C_ * H_ * WF_ * 2;   // stride between F branches

#if HAVE_ASYNC_LDS
    // async DMA of the image into LDS: rows packed at the head of each
    // padded float2 row slot; expanded to (re, 0) in place afterwards.
    for (int i = tid; i < HW_ / 4; i += 256) {       // 4096 x 16-byte chunks
        int h = i >> 5, j = i & 31;                  // 32 chunks per row
        char* ldst = (char*)S + (size_t)h * (PITCH * sizeof(float2)) + (size_t)j * 16;
        __builtin_amdgcn_global_load_async_to_lds_b128(
            to_glob_v4(img + (size_t)i * 4), to_lds_v4(ldst), 0, 0);
    }
    __builtin_amdgcn_s_wait_asynccnt(0);
    __syncthreads();
    if (tid < 128) {                                  // expand packed -> float2
        float* row = (float*)(S + (size_t)tid * PITCH);
        for (int w = 127; w >= 0; --w) {
            float v = row[w];
            row[2 * w]     = v;
            row[2 * w + 1] = 0.f;
        }
    }
#else
    for (int i = tid; i < HW_; i += 256) {
        int h = i >> 7, w = i & 127;
        S[h * PITCH + w] = make_float2(img[i], 0.f);
    }
#endif
    // stream the filter weights toward L2 while the forward FFT runs
    {
        const char* base = (const char*)dwc;
        const size_t span = (size_t)4 * fs * sizeof(float);
        for (size_t off = (size_t)tid * 128; off < span; off += 256 * 128)
            __builtin_prefetch(base + off, 0, 1);
    }
    __syncthreads();

    fft128_batch(S, TW, PITCH, 1, 0, tid);   // rows (along W)
    fft128_batch(S, TW, 1, PITCH, 0, tid);   // cols (along H)

    // interior columns 1..63: weighted = F^2 * G
    for (int i = tid; i < 128 * 63; i += 256) {
        int h = i / 63;
        int w = 1 + (i % 63);
        size_t o = (size_t)(h * WF_ + w) * 2;
        float gr = wt0*dwc[o]   + wt1*dwc[o+fs]   + wt2*dwc[o+2*fs]   + wt3*dwc[o+3*fs];
        float gi = wt0*dwc[o+1] + wt1*dwc[o+1+fs] + wt2*dwc[o+1+2*fs] + wt3*dwc[o+1+3*fs];
        float2 F = S[h * PITCH + w];
        float f2r = F.x * F.x - F.y * F.y;
        float f2i = 2.f * F.x * F.y;
        S[h * PITCH + w] = make_float2(f2r * gr - f2i * gi, f2r * gi + f2i * gr);
    }
    // boundary columns w=0 and w=64: Hermitian-projected filter response
    if (tid < 130) {
        int h = tid >> 1;
        int w = (tid & 1) ? 64 : 0;
        if (h <= 64) {
            int hm = (128 - h) & 127;
            size_t oa = (size_t)(h  * WF_ + w) * 2;
            size_t ob = (size_t)(hm * WF_ + w) * 2;
            float gar = wt0*dwc[oa]   + wt1*dwc[oa+fs]   + wt2*dwc[oa+2*fs]   + wt3*dwc[oa+3*fs];
            float gai = wt0*dwc[oa+1] + wt1*dwc[oa+1+fs] + wt2*dwc[oa+1+2*fs] + wt3*dwc[oa+1+3*fs];
            float gbr = wt0*dwc[ob]   + wt1*dwc[ob+fs]   + wt2*dwc[ob+2*fs]   + wt3*dwc[ob+3*fs];
            float gbi = wt0*dwc[ob+1] + wt1*dwc[ob+1+fs] + wt2*dwc[ob+1+2*fs] + wt3*dwc[ob+1+3*fs];
            float2 Fa = S[h  * PITCH + w];
            float2 Fb = S[hm * PITCH + w];
            float tr_ = Fa.x * gar - Fa.y * gai, ti_ = Fa.x * gai + Fa.y * gar;  // Fa*Ga
            float ur_ = Fb.x * gbr - Fb.y * gbi, ui_ = Fb.x * gbi + Fb.y * gbr;  // Fb*Gb
            float fr  = 0.5f * (tr_ + ur_), fi  = 0.5f * (ti_ - ui_);
            float fr2 = 0.5f * (ur_ + tr_), fi2 = 0.5f * (ui_ - ti_);
            S[h  * PITCH + w] = make_float2(Fa.x * fr  - Fa.y * fi,  Fa.x * fi  + Fa.y * fr);
            S[hm * PITCH + w] = make_float2(Fb.x * fr2 - Fb.y * fi2, Fb.x * fi2 + Fb.y * fr2);
        }
    }
    __syncthreads();

    // Hermitian extension to columns 65..127
    for (int i = tid; i < 128 * 63; i += 256) {
        int w = 65 + (i % 63);
        int h = i / 63;
        float2 v = S[((128 - h) & 127) * PITCH + (128 - w)];
        S[h * PITCH + w] = make_float2(v.x, -v.y);
    }
    __syncthreads();

    fft128_batch(S, TW, 1, PITCH, 1, tid);   // inverse cols
    fft128_batch(S, TW, PITCH, 1, 1, tid);   // inverse rows

    const float scale = 1.0f / (float)HW_;
    for (int i = tid; i < HW_; i += 256) {
        int h = i >> 7, w = i & 127;
        out[(size_t)bc * HW_ + i] = S[h * PITCH + w].x * scale + rw * img[i];
    }
}

// ===================================================================
// host-side launcher
// ===================================================================
extern "C" void kernel_launch(void* const* d_in, const int* in_sizes, int n_in,
                              void* d_out, int out_size, void* d_ws, size_t ws_size,
                              hipStream_t stream)
{
    const float* feat = (const float*)d_in[0];
    const float* gng  = (const float*)d_in[1];
    const float* gnb  = (const float*)d_in[2];
    const float* aggw = (const float*)d_in[3];
    const float* aggb = (const float*)d_in[4];
    const float* wgw  = (const float*)d_in[5];
    const float* wgb  = (const float*)d_in[6];
    const float* dsw  = (const float*)d_in[7];
    const float* rw   = (const float*)d_in[8];
    float* out = (float*)d_out;

    char* ws = (char*)d_ws;
    float*  mu     = (float*)(ws + 0);       // 128
    float*  rsig   = (float*)(ws + 512);     // 128
    float*  beff   = (float*)(ws + 1024);    // 64
    float*  pooled = (float*)(ws + 1280);    // 512
    float*  gatew  = (float*)(ws + 3328);    // 32
    bf16_t* Bw     = (bf16_t*)(ws + 3584);   // 4096 bf16

    zero_kernel<<<2, 256, 0, stream>>>(pooled, 512);
    gn_stats_kernel<<<128, 256, 0, stream>>>(feat, mu, rsig);
    prep_weights_kernel<<<1, 64, 0, stream>>>(aggw, aggb, gng, gnb, Bw, beff);
    gate_gemm_kernel<<<dim3(16, 8), 256, 0, stream>>>(feat, mu, rsig, Bw, beff, pooled);
    gate_softmax_kernel<<<8, 64, 0, stream>>>(pooled, wgw, wgb, gatew);

    const size_t shbytes = (size_t)(128 * PITCH + 64) * sizeof(float2);  // ~132.6 KB
    (void)hipFuncSetAttribute(reinterpret_cast<const void*>(&spectral_kernel),
                              hipFuncAttributeMaxDynamicSharedMemorySize, (int)shbytes);
    spectral_kernel<<<512, 256, shbytes, stream>>>(feat, dsw, gatew, rw, out);
}
